// MultiHeadSelfAttention_16604343567184
// MI455X (gfx1250) — compile-verified, matrix operand order checked by reading, not run on hardware
//
#include <hip/hip_runtime.h>
#include <hip/hip_bf16.h>

typedef __attribute__((ext_vector_type(16))) _Float16 v16h;
typedef __attribute__((ext_vector_type(8)))  _Float16 v8h;
typedef __attribute__((ext_vector_type(8)))  float    v8f;

#define D_MODEL 1024
#define HEADS   16
#define DHEAD   64
#define BATCH   4
#define SEQ     2048
#define MROWS   (BATCH * SEQ)   // 8192

// ---------------------------------------------------------------------------
// Fragment loader: 16x32 f16 tile from LDS, row stride `ld` (halves).
// Matches CDNA5 16-bit A-matrix layout (ISA 7.12.2): lane r in [0,16) holds
// row M=r; half 0 holds K 0-7 and 16-23, half 1 holds K 8-15 and 24-31.
// B matrix (K x N) uses the same pattern on an [N][K] tile (N -> rows).
// All addresses are 16B aligned -> lowers to 2x ds_load_b128 per lane.
// ---------------------------------------------------------------------------
__device__ __forceinline__ v16h load_frag16x32(const _Float16* base, int ld) {
  const int lane = threadIdx.x & 31;
  const int r    = lane & 15;
  const int hh   = lane >> 4;           // 0 or 1
  const _Float16* p = base + r * ld + hh * 8;
  v8h lo = *(const v8h*)(p);            // K = h*8 + 0..7
  v8h hi = *(const v8h*)(p + 16);       // K = 16 + h*8 + 0..7
  return __builtin_shufflevector(lo, hi, 0,1,2,3,4,5,6,7,
                                          8,9,10,11,12,13,14,15);
}

__device__ __forceinline__ v8f wmma_f16(v16h a, v16h b, v8f c) {
  // D = A(16x32) x B(32x16) + C, f32 accumulate
  return __builtin_amdgcn_wmma_f32_16x16x32_f16(
      false, a, false, b, (short)0, c, false, false);
}

// ---------------------------------------------------------------------------
// Precision conversion helpers
// ---------------------------------------------------------------------------
__global__ void f32_to_f16_kernel(const float* __restrict__ src,
                                  _Float16* __restrict__ dst, int n) {
  int i = blockIdx.x * blockDim.x + threadIdx.x;
  if (i < n) dst[i] = (_Float16)src[i];
}

// src [K][N] f32 row-major  ->  dst [N][K] f16 row-major (K contiguous)
__global__ void transpose_f32_to_f16_kernel(const float* __restrict__ src,
                                            _Float16* __restrict__ dst,
                                            int K, int N) {
  int i = blockIdx.x * blockDim.x + threadIdx.x;
  if (i < K * N) {
    int k = i / N;
    int n = i - k * N;
    dst[(size_t)n * K + k] = (_Float16)src[i];
  }
}

// ---------------------------------------------------------------------------
// WMMA GEMM:  C[M][N] = A[M][1024] * BT[N][1024]^T + bias
// Block tile 128x128, BK=32, 256 threads (8 waves, 2x4), wave tile 64x32.
// MODE 0: fp32 output (projection).
// MODE 1: split f16 Q/K/V [B,H,S,dh]; Q is pre-scaled by 1/sqrt(dh) so the
//         attention inner loop needs no scale multiplies.
// ---------------------------------------------------------------------------
template <int MODE>
__global__ __launch_bounds__(256) void gemm_wmma_kernel(
    const _Float16* __restrict__ A,    // [M][1024] f16
    const _Float16* __restrict__ BT,   // [N][1024] f16 (pre-transposed)
    const float*    __restrict__ bias, // [N]
    float*          __restrict__ outF, // MODE 0
    _Float16* __restrict__ qb, _Float16* __restrict__ kb,
    _Float16* __restrict__ vb,         // MODE 1
    int N) {
  constexpr int KD  = 1024;
  constexpr int LDT = 40;              // 32 + 8 pad, keeps 16B alignment
  __shared__ __align__(16) _Float16 As[128 * LDT];
  __shared__ __align__(16) _Float16 Bs[128 * LDT];

  const int tid    = threadIdx.x;
  const int wave   = tid >> 5;
  const int lane   = tid & 31;
  const int halfId = lane >> 4;
  const int nloc   = lane & 15;
  const int wm     = wave >> 2;        // 0..1
  const int wn     = wave & 3;         // 0..3
  const int m0     = wm * 64;
  const int n0     = wn * 32;
  const int blockM = blockIdx.y * 128;
  const int blockN = blockIdx.x * 128;

  // cooperative tile load mapping: 2 threads per row, 16 halves each
  const int lrow = tid >> 1;
  const int lcol = (tid & 1) * 16;
  const _Float16* Arow = A  + (size_t)(blockM + lrow) * KD + lcol;
  const _Float16* Brow = BT + (size_t)(blockN + lrow) * KD + lcol;
  _Float16* AsW = &As[lrow * LDT + lcol];
  _Float16* BsW = &Bs[lrow * LDT + lcol];

  v8f zero = {};
  v8f acc[4][2];
#pragma unroll
  for (int mt = 0; mt < 4; ++mt)
#pragma unroll
    for (int nt = 0; nt < 2; ++nt) acc[mt][nt] = zero;

  for (int k0 = 0; k0 < KD; k0 += 32) {
    uint4 a0 = *(const uint4*)(Arow + k0);
    uint4 a1 = *(const uint4*)(Arow + k0 + 8);
    uint4 b0 = *(const uint4*)(Brow + k0);
    uint4 b1 = *(const uint4*)(Brow + k0 + 8);
    if (k0 + 32 < KD) {                    // warm next tile (global_prefetch_b8)
      __builtin_prefetch(Arow + k0 + 32, 0, 3);
      __builtin_prefetch(Brow + k0 + 32, 0, 3);
    }
    __syncthreads();                       // previous tile fully consumed
    *(uint4*)(AsW)     = a0; *(uint4*)(AsW + 8) = a1;
    *(uint4*)(BsW)     = b0; *(uint4*)(BsW + 8) = b1;
    __syncthreads();

    v16h aF[4], bF[2];
#pragma unroll
    for (int mt = 0; mt < 4; ++mt)
      aF[mt] = load_frag16x32(&As[(m0 + mt * 16) * LDT], LDT);
#pragma unroll
    for (int nt = 0; nt < 2; ++nt)
      bF[nt] = load_frag16x32(&Bs[(n0 + nt * 16) * LDT], LDT);
#pragma unroll
    for (int mt = 0; mt < 4; ++mt)
#pragma unroll
      for (int nt = 0; nt < 2; ++nt)
        acc[mt][nt] = wmma_f16(aF[mt], bF[nt], acc[mt][nt]);
  }

  // Epilogue. C layout: VGPR g, lanes 0-15 -> (M=g, N=lane); 16-31 -> (M=8+g).
#pragma unroll
  for (int mt = 0; mt < 4; ++mt) {
#pragma unroll
    for (int nt = 0; nt < 2; ++nt) {
      v8f c = acc[mt][nt];
#pragma unroll
      for (int g = 0; g < 8; ++g) {
        int row = blockM + m0 + mt * 16 + g + halfId * 8;
        int col = blockN + n0 + nt * 16 + nloc;
        float val = c[g] + bias[col];
        if (MODE == 0) {
          outF[(size_t)row * N + col] = val;
        } else {
          int b     = row >> 11;       // / SEQ
          int s     = row & (SEQ - 1);
          int which = col >> 10;       // 0=q 1=k 2=v
          int d     = col & (D_MODEL - 1);
          int hh    = d >> 6;
          int di    = d & (DHEAD - 1);
          if (which == 0) val *= 0.125f;   // fold 1/sqrt(dh) into Q
          _Float16* dst = (which == 0) ? qb : (which == 1) ? kb : vb;
          dst[((size_t)((b * HEADS + hh) * SEQ + s)) * DHEAD + di] =
              (_Float16)val;
        }
      }
    }
  }
}

// ---------------------------------------------------------------------------
// Flash attention: one block per (b, h, 128-row Q tile), 8 waves; wave w owns
// rows [w*16, w*16+16). Online softmax, Bc = 64 key columns per step.
// Q arrives pre-scaled, so scores come straight out of the WMMA accumulators.
// ---------------------------------------------------------------------------
__global__ __launch_bounds__(256) void attn_kernel(
    const _Float16* __restrict__ Qg, const _Float16* __restrict__ Kg,
    const _Float16* __restrict__ Vg, _Float16* __restrict__ ctx) {
  constexpr int LD = 72;               // 64 + 8 pad
  __shared__ __align__(16) _Float16 Qs[128 * LD];
  __shared__ __align__(16) _Float16 Ks[64 * LD];
  __shared__ __align__(16) _Float16 Vt[64 * LD];       // transposed: [d][s]
  __shared__ __align__(16) _Float16 Ps[8 * 16 * LD];   // per-wave P tile

  const int tid    = threadIdx.x;
  const int wave   = tid >> 5;
  const int lane   = tid & 31;
  const int halfId = lane >> 4;
  const int nloc   = lane & 15;

  const int tilesPerBH = SEQ / 128;    // 16
  const int bh   = blockIdx.x / tilesPerBH;
  const int tile = blockIdx.x - bh * tilesPerBH;
  const int b    = bh / HEADS;
  const int h    = bh - b * HEADS;
  const int row0 = tile * 128;
  const size_t headBase = (size_t)bh * SEQ * DHEAD;

  // Load Q block (128 x 64 halves): 2 threads per row, 32 halves each
  {
    int r = tid >> 1;
    int c = (tid & 1) * 32;
    const _Float16* src = Qg + headBase + (size_t)(row0 + r) * DHEAD + c;
    uint4 q0 = *(const uint4*)(src);
    uint4 q1 = *(const uint4*)(src + 8);
    uint4 q2 = *(const uint4*)(src + 16);
    uint4 q3 = *(const uint4*)(src + 24);
    _Float16* dstp = &Qs[r * LD + c];
    *(uint4*)(dstp)      = q0;
    *(uint4*)(dstp + 8)  = q1;
    *(uint4*)(dstp + 16) = q2;
    *(uint4*)(dstp + 24) = q3;
  }

  v8f zero = {};
  v8f O[4];
  float m_run[8], l_run[8];
#pragma unroll
  for (int dt = 0; dt < 4; ++dt) O[dt] = zero;
#pragma unroll
  for (int g = 0; g < 8; ++g) { m_run[g] = -1e30f; l_run[g] = 0.0f; }

  _Float16* Pw = &Ps[wave * 16 * LD];
  const _Float16* qbase = &Qs[(wave * 16) * LD];

  for (int j = 0; j < SEQ; j += 64) {
    __syncthreads();                   // previous K/V tile fully consumed
    {                                  // K tile: 64 x 64, [s][d] as-is
      int r = tid >> 2;
      int c = (tid & 3) * 16;
      const _Float16* src = Kg + headBase + (size_t)(j + r) * DHEAD + c;
      uint4 k0v = *(const uint4*)(src);
      uint4 k1v = *(const uint4*)(src + 8);
      if (j + 64 < SEQ) {              // warm next tiles while we compute
        __builtin_prefetch(src + 64 * DHEAD, 0, 3);
        __builtin_prefetch(Vg + headBase + (size_t)(j + 64 + r) * DHEAD + c,
                           0, 3);
      }
      _Float16* dstp = &Ks[r * LD + c];
      *(uint4*)(dstp)     = k0v;
      *(uint4*)(dstp + 8) = k1v;
    }
    for (int idx = tid; idx < 64 * 64; idx += 256) {   // V transposed
      int r = idx >> 6;
      int d = idx & 63;
      Vt[d * LD + r] = Vg[headBase + (size_t)(j + r) * DHEAD + d];
    }
    __syncthreads();

    // S = Q K^T : A = Q rows, B-frag source = Ks[[n][k]] directly
    v8f sAcc[4];
#pragma unroll
    for (int nt = 0; nt < 4; ++nt) sAcc[nt] = zero;
#pragma unroll
    for (int ks = 0; ks < 2; ++ks) {
      v16h aF = load_frag16x32(qbase + ks * 32, LD);
#pragma unroll
      for (int nt = 0; nt < 4; ++nt) {
        v16h bF = load_frag16x32(&Ks[(nt * 16) * LD] + ks * 32, LD);
        sAcc[nt] = wmma_f16(aF, bF, sAcc[nt]);
      }
    }

    // Online softmax. Row (g + halfId*8) lives in one 16-lane half ->
    // shfl_xor masks 1/2/4/8 reduce across its 16 column lanes only.
#pragma unroll
    for (int g = 0; g < 8; ++g) {
      float mx = -1e30f;
#pragma unroll
      for (int nt = 0; nt < 4; ++nt) mx = fmaxf(mx, sAcc[nt][g]);
      mx = fmaxf(mx, __shfl_xor(mx, 1, 32));
      mx = fmaxf(mx, __shfl_xor(mx, 2, 32));
      mx = fmaxf(mx, __shfl_xor(mx, 4, 32));
      mx = fmaxf(mx, __shfl_xor(mx, 8, 32));
      float newm = fmaxf(m_run[g], mx);
      float corr = __expf(m_run[g] - newm);
      m_run[g]   = newm;
      float psum = 0.0f;
#pragma unroll
      for (int nt = 0; nt < 4; ++nt) {
        float p = __expf(sAcc[nt][g] - newm);
        psum += p;
        Pw[(g + halfId * 8) * LD + nt * 16 + nloc] = (_Float16)p;
      }
      psum += __shfl_xor(psum, 1, 32);
      psum += __shfl_xor(psum, 2, 32);
      psum += __shfl_xor(psum, 4, 32);
      psum += __shfl_xor(psum, 8, 32);
      l_run[g] = l_run[g] * corr + psum;
#pragma unroll
      for (int dt = 0; dt < 4; ++dt) O[dt][g] *= corr;
    }

    // O += P V  (DS ops are in-order per wave: P stores precede these loads)
#pragma unroll
    for (int ks = 0; ks < 2; ++ks) {
      v16h pF = load_frag16x32(Pw + ks * 32, LD);
#pragma unroll
      for (int dt = 0; dt < 4; ++dt) {
        v16h bF = load_frag16x32(&Vt[(dt * 16) * LD] + ks * 32, LD);
        O[dt] = wmma_f16(pF, bF, O[dt]);
      }
    }
  }

  // Epilogue: normalize, write ctx f16 [B, S, D] with head interleave
#pragma unroll
  for (int dt = 0; dt < 4; ++dt) {
#pragma unroll
    for (int g = 0; g < 8; ++g) {
      int srow  = row0 + wave * 16 + g + halfId * 8;
      float val = O[dt][g] / l_run[g];
      ctx[((size_t)(b * SEQ + srow)) * D_MODEL + h * DHEAD + dt * 16 + nloc] =
          (_Float16)val;
    }
  }
}

// ---------------------------------------------------------------------------
// Launch: convert -> QKV GEMM -> flash attention -> projection GEMM
// Workspace layout (needs ~88 MB):
//   X16 16MB | WqkvT 6MB | WprojT 2MB | Q 16MB | K 16MB | V 16MB | ctx 16MB
// ---------------------------------------------------------------------------
extern "C" void kernel_launch(void* const* d_in, const int* in_sizes, int n_in,
                              void* d_out, int out_size, void* d_ws,
                              size_t ws_size, hipStream_t stream) {
  const float* x      = (const float*)d_in[0];
  const float* w_qkv  = (const float*)d_in[1];
  const float* b_qkv  = (const float*)d_in[2];
  const float* w_proj = (const float*)d_in[3];
  const float* b_proj = (const float*)d_in[4];
  float* out = (float*)d_out;

  char* ws = (char*)d_ws;
  size_t off = 0;
  _Float16* X16    = (_Float16*)(ws + off); off += (size_t)MROWS * D_MODEL * 2;
  _Float16* WqkvT  = (_Float16*)(ws + off); off += (size_t)3 * D_MODEL * D_MODEL * 2;
  _Float16* WprojT = (_Float16*)(ws + off); off += (size_t)D_MODEL * D_MODEL * 2;
  _Float16* Qb     = (_Float16*)(ws + off); off += (size_t)MROWS * D_MODEL * 2;
  _Float16* Kb     = (_Float16*)(ws + off); off += (size_t)MROWS * D_MODEL * 2;
  _Float16* Vb     = (_Float16*)(ws + off); off += (size_t)MROWS * D_MODEL * 2;
  _Float16* Ctx16  = (_Float16*)(ws + off); off += (size_t)MROWS * D_MODEL * 2;

  const int nX = MROWS * D_MODEL;                // 8388608
  f32_to_f16_kernel<<<dim3((nX + 255) / 256), 256, 0, stream>>>(x, X16, nX);

  const int nWq = D_MODEL * 3 * D_MODEL;
  transpose_f32_to_f16_kernel<<<dim3((nWq + 255) / 256), 256, 0, stream>>>(
      w_qkv, WqkvT, D_MODEL, 3 * D_MODEL);
  const int nWp = D_MODEL * D_MODEL;
  transpose_f32_to_f16_kernel<<<dim3((nWp + 255) / 256), 256, 0, stream>>>(
      w_proj, WprojT, D_MODEL, D_MODEL);

  gemm_wmma_kernel<1><<<dim3(3 * D_MODEL / 128, MROWS / 128), 256, 0, stream>>>(
      X16, WqkvT, b_qkv, nullptr, Qb, Kb, Vb, 3 * D_MODEL);

  attn_kernel<<<dim3(BATCH * HEADS * (SEQ / 128)), 256, 0, stream>>>(
      Qb, Kb, Vb, Ctx16);

  gemm_wmma_kernel<0><<<dim3(D_MODEL / 128, MROWS / 128), 256, 0, stream>>>(
      Ctx16, WprojT, b_proj, out, nullptr, nullptr, nullptr, D_MODEL);
}
